// SimpleTransformer_56796647522824
// MI455X (gfx1250) — matrix operand order repaired and checked by reading, not compiled, not run on hardware
//
#include <hip/hip_runtime.h>

// ---------------------------------------------------------------------------
// Types for CDNA5 WMMA (wave32): v_wmma_f32_16x16x32_bf16
// ---------------------------------------------------------------------------
typedef __attribute__((ext_vector_type(16))) __bf16 bf16x16;
typedef __attribute__((ext_vector_type(8)))  __bf16 bf16x8;
typedef __attribute__((ext_vector_type(8)))  float  f32x8;

// fp32 -> bf16 round-to-nearest-even
__device__ __forceinline__ unsigned short f2bf(float x) {
    unsigned int u = __float_as_uint(x);
    u += 0x7FFFu + ((u >> 16) & 1u);
    return (unsigned short)(u >> 16);
}
__device__ __forceinline__ unsigned int pack2bf(float a, float b) {
    return (unsigned int)f2bf(a) | ((unsigned int)f2bf(b) << 16);
}

// Assemble a 16-element bf16 fragment from two contiguous 8-element (16B) LDS groups.
__device__ __forceinline__ bf16x16 load_frag(const unsigned short* p0,
                                             const unsigned short* p1) {
    bf16x8 lo = *reinterpret_cast<const bf16x8*>(p0);
    bf16x8 hi = *reinterpret_cast<const bf16x8*>(p1);
    bf16x16 r;
#pragma unroll
    for (int i = 0; i < 8; ++i) { r[i] = lo[i]; r[i + 8] = hi[i]; }
    return r;
}

__device__ __forceinline__ f32x8 wmma_bf16(bf16x16 a, bf16x16 b, f32x8 c) {
    return __builtin_amdgcn_wmma_f32_16x16x32_bf16(false, a, false, b,
                                                   (short)0, c, false, false);
}

// ---------------------------------------------------------------------------
// GEMM: C[M,N] = A[M,K] @ W[N,K]^T + bias[N]  (optional ReLU)
// Block: 256 threads (8 waves), tile 128x128, K step 32 via LDS (bf16).
// Wave w: rows (w&3)*32..+31 (2 M-tiles), cols (w>>2)*64..+63 (4 N-tiles).
// ---------------------------------------------------------------------------
template <bool RELU>
__global__ __launch_bounds__(256) void gemm_bias_kernel(
    const float* __restrict__ A, const float* __restrict__ W,
    const float* __restrict__ bias, float* __restrict__ C,
    int M, int N, int K)
{
    __shared__ unsigned short Alds[128][32];
    __shared__ unsigned short Blds[128][32];

    const int tid  = threadIdx.x;
    const int lane = tid & 31;
    const int wave = tid >> 5;
    const int ln   = lane & 15;
    const int half = (lane >> 4) & 1;
    const int mbase = (wave & 3) * 32;
    const int nbase = (wave >> 2) * 64;
    const int m0 = blockIdx.y * 128;
    const int n0 = blockIdx.x * 128;

    f32x8 acc[2][4];
#pragma unroll
    for (int mt = 0; mt < 2; ++mt)
#pragma unroll
        for (int nt = 0; nt < 4; ++nt) { f32x8 z = {}; acc[mt][nt] = z; }

    for (int k0 = 0; k0 < K; k0 += 32) {
        // Stage A[128][32] and W[128][32] tiles as bf16
#pragma unroll
        for (int j = 0; j < 4; ++j) {
            int i   = tid + j * 256;     // 0..1023 float4 slots
            int row = i >> 3;
            int c4  = (i & 7) << 2;
            float4 fa = *reinterpret_cast<const float4*>(
                A + (size_t)(m0 + row) * K + k0 + c4);
            uint2 pa; pa.x = pack2bf(fa.x, fa.y); pa.y = pack2bf(fa.z, fa.w);
            *reinterpret_cast<uint2*>(&Alds[row][c4]) = pa;
            float4 fb = *reinterpret_cast<const float4*>(
                W + (size_t)(n0 + row) * K + k0 + c4);
            uint2 pb; pb.x = pack2bf(fb.x, fb.y); pb.y = pack2bf(fb.z, fb.w);
            *reinterpret_cast<uint2*>(&Blds[row][c4]) = pb;
        }
        __syncthreads();

        bf16x16 afrag[2], bfrag[4];
#pragma unroll
        for (int mt = 0; mt < 2; ++mt) {
            const unsigned short* rp = &Alds[mbase + mt * 16 + ln][0];
            afrag[mt] = load_frag(rp + half * 8, rp + 16 + half * 8);
        }
#pragma unroll
        for (int nt = 0; nt < 4; ++nt) {
            const unsigned short* rp = &Blds[nbase + nt * 16 + ln][0];
            bfrag[nt] = load_frag(rp + half * 8, rp + 16 + half * 8);
        }
#pragma unroll
        for (int mt = 0; mt < 2; ++mt)
#pragma unroll
            for (int nt = 0; nt < 4; ++nt)
                acc[mt][nt] = wmma_bf16(afrag[mt], bfrag[nt], acc[mt][nt]);
        __syncthreads();
    }

    // Epilogue: bias (+ReLU), store fp32
#pragma unroll
    for (int nt = 0; nt < 4; ++nt) {
        const int col = n0 + nbase + nt * 16 + ln;
        const float bv = bias ? bias[col] : 0.0f;
#pragma unroll
        for (int mt = 0; mt < 2; ++mt) {
#pragma unroll
            for (int r = 0; r < 8; ++r) {
                float v = acc[mt][nt][r] + bv;
                if (RELU) v = fmaxf(v, 0.0f);
                const int row = m0 + mbase + mt * 16 + r + half * 8;
                C[(size_t)row * N + col] = v;
            }
        }
    }
}

// ---------------------------------------------------------------------------
// Flash attention over Q/K/V laid out as [B*S, H*64] fp32.
// Block: 256 threads, handles one (b, h) and 128 query rows (16 per wave).
// K/V streamed in 128-key LDS tiles; online softmax over 32-key chunks.
// The causal chunk limit is computed in SGPRs (readfirstlane) so the inner
// loop is wave-uniform: no exec masking around WMMA, no accumulator phi copies.
// ---------------------------------------------------------------------------
__global__ __launch_bounds__(256) void flash_attn_kernel(
    const float* __restrict__ Q, const float* __restrict__ K,
    const float* __restrict__ V, float* __restrict__ O,
    int S, int causal)
{
    __shared__ unsigned short Qlds[128][64];
    __shared__ unsigned short Klds[128][64];
    __shared__ unsigned short Vlds[64][128];   // transposed: [d][key]
    __shared__ unsigned short Plds[8][16][32]; // per-wave P scratch

    const int tid  = threadIdx.x;
    const int lane = tid & 31;
    const int wave = tid >> 5;
    const int ln   = lane & 15;
    const int half = (lane >> 4) & 1;
    const int h  = blockIdx.y;
    const int b  = blockIdx.z;
    const int qb = blockIdx.x * 128;
    const size_t ldx = 512;  // H*64

    // Stage Q tile [128][64] as bf16
#pragma unroll
    for (int j = 0; j < 8; ++j) {
        int i = tid + j * 256;           // 0..2047 float4 slots
        int row = i >> 4;
        int c4  = (i & 15) << 2;
        float4 f = *reinterpret_cast<const float4*>(
            Q + (size_t)(b * S + qb + row) * ldx + h * 64 + c4);
        uint2 p; p.x = pack2bf(f.x, f.y); p.y = pack2bf(f.z, f.w);
        *reinterpret_cast<uint2*>(&Qlds[row][c4]) = p;
    }
    __syncthreads();

    // Per-wave Q A-fragments for the two 32-wide d chunks
    bf16x16 qa[2];
#pragma unroll
    for (int dc = 0; dc < 2; ++dc) {
        const unsigned short* rp = &Qlds[wave * 16 + ln][dc * 32];
        qa[dc] = load_frag(rp + half * 8, rp + 16 + half * 8);
    }

    f32x8 acc[4];
#pragma unroll
    for (int nt = 0; nt < 4; ++nt) { f32x8 z = {}; acc[nt] = z; }
    float mrow[8], lrow[8];
#pragma unroll
    for (int r = 0; r < 8; ++r) { mrow[r] = -INFINITY; lrow[r] = 0.0f; }

    const int qw  = qb + wave * 16;  // this wave's first query row (global)
    // Wave-uniform copy in an SGPR: the compiler can't prove tid>>5 is uniform,
    // so force it scalar to get uniform (s_cbranch) inner-loop control flow.
    const int qwu = __builtin_amdgcn_readfirstlane(qw);
    const int kend = causal ? (qb + 128) : S;
    const float scale = 0.125f;      // 1/sqrt(64)

    for (int kb = 0; kb < kend; kb += 128) {
        __syncthreads();  // previous tile's readers done before restage
        // Stage K tile [128][64] bf16
#pragma unroll
        for (int j = 0; j < 8; ++j) {
            int i = tid + j * 256;
            int row = i >> 4;
            int c4  = (i & 15) << 2;
            float4 f = *reinterpret_cast<const float4*>(
                K + (size_t)(b * S + kb + row) * ldx + h * 64 + c4);
            uint2 p; p.x = pack2bf(f.x, f.y); p.y = pack2bf(f.z, f.w);
            *reinterpret_cast<uint2*>(&Klds[row][c4]) = p;
        }
        // Stage V tile transposed: Vlds[d][key], packing key pairs.
        // Consecutive lanes read consecutive d of the same key row: coalesced.
#pragma unroll
        for (int j = 0; j < 16; ++j) {
            int i  = tid + j * 256;      // 0..4095 (64 d x 64 key-pairs)
            int d  = i & 63;
            int kk = (i >> 6) * 2;
            const float v0 = V[(size_t)(b * S + kb + kk)     * ldx + h * 64 + d];
            const float v1 = V[(size_t)(b * S + kb + kk + 1) * ldx + h * 64 + d];
            *reinterpret_cast<unsigned int*>(&Vlds[d][kk]) = pack2bf(v0, v1);
        }
        __syncthreads();

        // Wave-uniform chunk count: chunk c needs kb + 32c <= qw + 15.
        int cmax = 4;
        if (causal) {
            int lim = ((qwu + 15 - kb) >> 5) + 1;   // kb <= qb <= qwu here
            cmax = lim < 4 ? lim : 4;
        }

        for (int c = 0; c < cmax; ++c) {
            const int ck = c * 32;

            // Scores for two adjacent 16-key tiles (keys kb+ck .. kb+ck+31)
            f32x8 s0 = {}, s1 = {};
            {
                const unsigned short* kp = &Klds[ck + ln][half * 8];
                s0 = wmma_bf16(qa[0], load_frag(kp, kp + 16), s0);
                s0 = wmma_bf16(qa[1], load_frag(kp + 32, kp + 48), s0);
            }
            {
                const unsigned short* kp = &Klds[ck + 16 + ln][half * 8];
                s1 = wmma_bf16(qa[0], load_frag(kp, kp + 16), s1);
                s1 = wmma_bf16(qa[1], load_frag(kp + 32, kp + 48), s1);
            }
#pragma unroll
            for (int r = 0; r < 8; ++r) {
                s0[r] = s0[r] * scale;
                s1[r] = s1[r] * scale;
            }
            if (causal) {
                const int kg0 = kb + ck + ln;
#pragma unroll
                for (int r = 0; r < 8; ++r) {
                    const int qg = qw + r + half * 8;
                    s0[r] = (kg0 > qg)      ? -INFINITY : s0[r];
                    s1[r] = (kg0 + 16 > qg) ? -INFINITY : s1[r];
                }
            }
            // Online softmax across this 32-key chunk (row reductions over 16 lanes)
#pragma unroll
            for (int r = 0; r < 8; ++r) {
                float v = fmaxf(s0[r], s1[r]);
#pragma unroll
                for (int off = 8; off >= 1; off >>= 1)
                    v = fmaxf(v, __shfl_xor(v, off, 16));
                const float nm    = fmaxf(mrow[r], v);
                const float alpha = __expf(mrow[r] - nm);
                mrow[r] = nm;
                const float p0 = __expf(s0[r] - nm);
                const float p1 = __expf(s1[r] - nm);
                float ps = p0 + p1;
#pragma unroll
                for (int off = 8; off >= 1; off >>= 1)
                    ps += __shfl_xor(ps, off, 16);
                lrow[r] = lrow[r] * alpha + ps;
#pragma unroll
                for (int nt = 0; nt < 4; ++nt)
                    acc[nt][r] = acc[nt][r] * alpha;
                Plds[wave][r + half * 8][ln]      = f2bf(p0);
                Plds[wave][r + half * 8][16 + ln] = f2bf(p1);
            }
            // O += P @ V  (A = P[16x32], B = V^T fragments from Vlds[d][key])
            {
                const unsigned short* pp = &Plds[wave][ln][0];
                bf16x16 pa = load_frag(pp + half * 8, pp + 16 + half * 8);
#pragma unroll
                for (int nt = 0; nt < 4; ++nt) {
                    const unsigned short* vp = &Vlds[nt * 16 + ln][ck + half * 8];
                    acc[nt] = wmma_bf16(pa, load_frag(vp, vp + 16), acc[nt]);
                }
            }
        }
    }

    // Normalize and store O in [B*S, H*64] fp32
#pragma unroll
    for (int nt = 0; nt < 4; ++nt) {
        const int col = h * 64 + nt * 16 + ln;
#pragma unroll
        for (int r = 0; r < 8; ++r) {
            const int row = qb + wave * 16 + r + half * 8;
            O[(size_t)(b * S + row) * ldx + col] = acc[nt][r] / lrow[r];
        }
    }
}

// ---------------------------------------------------------------------------
// out[row] = LayerNorm(res[row] + c[row]) * g + beta     (E = 512, eps = 1e-15)
// One 256-thread block per row; 2 elements per thread.
// ---------------------------------------------------------------------------
__global__ __launch_bounds__(256) void add_layernorm_kernel(
    const float* __restrict__ res, const float* __restrict__ c,
    const float* __restrict__ g, const float* __restrict__ beta,
    float* __restrict__ out)
{
    const int row = blockIdx.x;
    const int tid = threadIdx.x;
    const size_t base = (size_t)row * 512;
    const float x0 = res[base + tid]       + c[base + tid];
    const float x1 = res[base + tid + 256] + c[base + tid + 256];
    float s = x0 + x1;
    float q = x0 * x0 + x1 * x1;
#pragma unroll
    for (int off = 16; off >= 1; off >>= 1) {
        s += __shfl_xor(s, off, 32);
        q += __shfl_xor(q, off, 32);
    }
    __shared__ float ss[8], sq[8];
    if ((tid & 31) == 0) { ss[tid >> 5] = s; sq[tid >> 5] = q; }
    __syncthreads();
    float ts = 0.0f, tq = 0.0f;
#pragma unroll
    for (int i = 0; i < 8; ++i) { ts += ss[i]; tq += sq[i]; }
    const float mean = ts * (1.0f / 512.0f);
    const float var  = tq * (1.0f / 512.0f) - mean * mean;
    const float inv  = rsqrtf(var + 1e-15f);
    out[base + tid]       = g[tid]       * (x0 - mean) * inv + beta[tid];
    out[base + tid + 256] = g[tid + 256] * (x1 - mean) * inv + beta[tid + 256];
}

// ---------------------------------------------------------------------------
// Launch: full decoder layer pipeline
// ---------------------------------------------------------------------------
extern "C" void kernel_launch(void* const* d_in, const int* in_sizes, int n_in,
                              void* d_out, int out_size, void* d_ws, size_t ws_size,
                              hipStream_t stream)
{
    (void)in_sizes; (void)n_in; (void)out_size; (void)ws_size;

    const float* dec = (const float*)d_in[0];
    const float* enc = (const float*)d_in[1];
    const float* Wq1 = (const float*)d_in[2];  const float* bq1 = (const float*)d_in[3];
    const float* Wk1 = (const float*)d_in[4];  const float* bk1 = (const float*)d_in[5];
    const float* Wv1 = (const float*)d_in[6];  const float* bv1 = (const float*)d_in[7];
    const float* Wo1 = (const float*)d_in[8];  const float* bo1 = (const float*)d_in[9];
    const float* Wq2 = (const float*)d_in[10]; const float* bq2 = (const float*)d_in[11];
    const float* Wk2 = (const float*)d_in[12]; const float* bk2 = (const float*)d_in[13];
    const float* Wv2 = (const float*)d_in[14]; const float* bv2 = (const float*)d_in[15];
    const float* Wo2 = (const float*)d_in[16]; const float* bo2 = (const float*)d_in[17];
    const float* Wf1 = (const float*)d_in[18]; const float* bf1 = (const float*)d_in[19];
    const float* Wf2 = (const float*)d_in[20]; const float* bf2 = (const float*)d_in[21];
    const float* g1  = (const float*)d_in[22]; const float* be1 = (const float*)d_in[23];
    const float* g2  = (const float*)d_in[24]; const float* be2 = (const float*)d_in[25];
    const float* g3  = (const float*)d_in[26]; const float* be3 = (const float*)d_in[27];

    const int S = 1024, E = 512, HID = 2048;
    const int M = 8 * S;                 // B*S rows
    const size_t U = (size_t)M * E;      // elements per [M,E] buffer

    float* ws  = (float*)d_ws;
    float* qb_ = ws;                     // Q projections
    float* kb_ = ws + 1 * U;             // K projections
    float* vb_ = ws + 2 * U;             // V projections
    float* ob_ = ws + 3 * U;             // attention output / x2
    float* pb_ = ws + 4 * U;             // pre-LN projection output
    float* x1_ = ws + 5 * U;             // x1
    float* hb_ = ws + 6 * U;             // FFN hidden [M, HID]

    dim3 blk(256);
    dim3 gE(E / 128, M / 128);
    dim3 gH(HID / 128, M / 128);
    dim3 gattn(S / 128, 8, 8);
    dim3 gln(M);

    // --- Masked self-attention block ---
    gemm_bias_kernel<false><<<gE, blk, 0, stream>>>(dec, Wq1, bq1, qb_, M, E, E);
    gemm_bias_kernel<false><<<gE, blk, 0, stream>>>(dec, Wk1, bk1, kb_, M, E, E);
    gemm_bias_kernel<false><<<gE, blk, 0, stream>>>(dec, Wv1, bv1, vb_, M, E, E);
    flash_attn_kernel<<<gattn, blk, 0, stream>>>(qb_, kb_, vb_, ob_, S, 1);
    gemm_bias_kernel<false><<<gE, blk, 0, stream>>>(ob_, Wo1, bo1, pb_, M, E, E);
    add_layernorm_kernel<<<gln, blk, 0, stream>>>(dec, pb_, g1, be1, x1_);

    // --- Cross-attention block ---
    gemm_bias_kernel<false><<<gE, blk, 0, stream>>>(x1_, Wq2, bq2, qb_, M, E, E);
    gemm_bias_kernel<false><<<gE, blk, 0, stream>>>(enc, Wk2, bk2, kb_, M, E, E);
    gemm_bias_kernel<false><<<gE, blk, 0, stream>>>(enc, Wv2, bv2, vb_, M, E, E);
    flash_attn_kernel<<<gattn, blk, 0, stream>>>(qb_, kb_, vb_, ob_, S, 0);
    gemm_bias_kernel<false><<<gE, blk, 0, stream>>>(ob_, Wo2, bo2, pb_, M, E, E);
    float* x2_ = ob_;  // ob_ free after the GEMM above
    add_layernorm_kernel<<<gln, blk, 0, stream>>>(x1_, pb_, g2, be2, x2_);

    // --- Feedforward block ---
    gemm_bias_kernel<true ><<<gH, blk, 0, stream>>>(x2_, Wf1, bf1, hb_, M, HID, E);
    gemm_bias_kernel<false><<<gE, blk, 0, stream>>>(hb_, Wf2, bf2, pb_, M, E, HID);
    add_layernorm_kernel<<<gln, blk, 0, stream>>>(x2_, pb_, g3, be3, (float*)d_out);
}
